// PN2PartSegMsgEncoder_41532333752454
// MI455X (gfx1250) — matrix operand order repaired and checked by reading, not compiled, not run on hardware
//
#include <hip/hip_runtime.h>
#include <hip/hip_bf16.h>

// ---------------------------------------------------------------------------
// PointNet++ MSG encoder for MI455X (gfx1250, wave32, WMMA).
// All MLP layers run through v_wmma_f32_16x16x32_f16 (f16 in, f32 acc).
// Weights are pre-packed once into the WMMA B-fragment layout (f16).
// GEMM tile: 64x64 per block (4 waves); A fragments are preloaded so the
// 4 WMMAs per K-step issue back-to-back behind a single s_wait_dscnt.
// ---------------------------------------------------------------------------

typedef __attribute__((ext_vector_type(16))) _Float16 v16h;
typedef __attribute__((ext_vector_type(4)))  _Float16 v4h;
typedef __attribute__((ext_vector_type(8)))  float    v8f;

#define WMMA_F16(a, b, c) \
  __builtin_amdgcn_wmma_f32_16x16x32_f16(false, (a), false, (b), (short)0, (c), false, false)

#define AS_STRIDE 36  // halfs per LDS row (72B: 8B-aligned quads, bank-skewed)

// ---------------------------------------------------------------------------
// Pre-pack W[CO,CI] (f32, row-major) into B-fragment tiles (f16):
//   Bp[((kt*NT + nt)*32 + lane)*16 + j] = W[nt*16 + (lane&15)][kt*32 + 16*(lane>>4) + j]
// so each wave's B fragment for tile (kt,nt) is one contiguous 32B load/lane.
// ---------------------------------------------------------------------------
__global__ void pack_w_kernel(const float* __restrict__ W, _Float16* __restrict__ Bp,
                              int CO, int CI, int NT, int KT)
{
  int i = blockIdx.x * blockDim.x + threadIdx.x;
  if (i >= NT * KT * 512) return;
  int j    = i & 15;
  int lane = (i >> 4) & 31;
  int t    = i >> 9;                 // kt*NT + nt
  int nt   = t % NT, kt = t / NT;
  int n = nt * 16 + (lane & 15);
  int k = kt * 32 + ((lane >> 4) << 4) + j;
  float v = 0.0f;
  if (n < CO && k < CI) v = W[(size_t)n * CI + k];
  Bp[i] = (_Float16)v;
}

// Build one A fragment (row = mt*16 + nn) from the LDS-staged 64x32 tile.
__device__ __forceinline__ v16h a_frag_from_lds(const _Float16* As, int mt, int nn, int half)
{
  const int rbase = (mt * 16 + nn) * AS_STRIDE;
  v16h a;
#pragma unroll
  for (int v = 0; v < 8; ++v) {
    int kb = (half << 4) + (v << 1); // A layout: k = 16*half + s (matches packed B)
    a[2 * v]     = As[rbase + kb];
    a[2 * v + 1] = As[rbase + kb + 1];
  }
  return a;
}

// ---------------------------------------------------------------------------
// Y[M,CO] = act(X[M,CI] @ W^T + bias), W pre-packed as Bp. 64x64 block tile.
// ---------------------------------------------------------------------------
__global__ __launch_bounds__(128)
void linear_wmma_kernel(const float* __restrict__ X, const _Float16* __restrict__ Bp,
                        const float* __restrict__ bias, float* __restrict__ Y,
                        int M, int CI, int CO, int NT, int relu)
{
  __shared__ _Float16 As[64 * AS_STRIDE];
  const int lane  = threadIdx.x & 31;
  const int wave  = threadIdx.x >> 5;
  const int m0    = blockIdx.x * 64;
  const int nt    = blockIdx.y * 4 + wave;   // this wave's N tile
  const int n0    = nt * 16;
  const int half  = lane >> 4;
  const int nn    = lane & 15;
  const bool nok  = (nt < NT);
  const v16h* BpV = (const v16h*)Bp;

  v8f acc[4] = {};
  const int KT = (CI + 31) >> 5;
  const bool mfull = (m0 + 64 <= M);
  const bool civec = ((CI & 3) == 0);

  for (int kt = 0; kt < KT; ++kt) {
    const int k0 = kt << 5;
    // ---- stage 64x32 A tile (f32 -> f16) into LDS ------------------------
    if (mfull && civec && (k0 + 32 <= CI)) {
      // fast path: vectorized, branch-free
      for (int q = threadIdx.x; q < 64 * 8; q += 128) {
        int r = q >> 3, cq = (q & 7) << 2;
        const float4 f = *(const float4*)(X + (size_t)(m0 + r) * CI + k0 + cq);
        v4h h; h.x = (_Float16)f.x; h.y = (_Float16)f.y;
               h.z = (_Float16)f.z; h.w = (_Float16)f.w;
        *(v4h*)&As[r * AS_STRIDE + cq] = h;
      }
    } else {
      for (int e = threadIdx.x; e < 64 * 32; e += 128) {
        int r = e >> 5, c = e & 31;
        int gm = m0 + r, gk = k0 + c;
        float v = 0.0f;
        if (gm < M && gk < CI) v = X[(size_t)gm * CI + gk];
        As[r * AS_STRIDE + c] = (_Float16)v;
      }
    }
    __syncthreads();

    // ---- one B fragment per wave (issued first: loadcnt overlaps dscnt) --
    v16h b = {};
    if (nok) {
      b = BpV[(size_t)(kt * NT + nt) * 32 + lane];
      if (kt + 1 < KT)
        __builtin_prefetch(&BpV[(size_t)((kt + 1) * NT + nt) * 32 + lane], 0, 1);
    }

    // ---- preload all 4 A fragments, then 4 back-to-back WMMAs ------------
    v16h a0 = a_frag_from_lds(As, 0, nn, half);
    v16h a1 = a_frag_from_lds(As, 1, nn, half);
    v16h a2 = a_frag_from_lds(As, 2, nn, half);
    v16h a3 = a_frag_from_lds(As, 3, nn, half);
    acc[0] = WMMA_F16(a0, b, acc[0]);
    acc[1] = WMMA_F16(a1, b, acc[1]);
    acc[2] = WMMA_F16(a2, b, acc[2]);
    acc[3] = WMMA_F16(a3, b, acc[3]);
    __syncthreads();
  }

  // ---- epilogue: C layout = VGPR r -> row r(+8 for lanes 16-31), col nn --
  const int gn = n0 + nn;
  const float bv = (gn < CO && bias) ? bias[gn] : 0.0f;
#pragma unroll
  for (int mt = 0; mt < 4; ++mt) {
#pragma unroll
    for (int r = 0; r < 8; ++r) {
      int gm = m0 + mt * 16 + r + (half << 3);
      if (gm < M && gn < CO) {
        float v = acc[mt][r] + bv;
        if (relu) v = fmaxf(v, 0.0f);
        Y[(size_t)gm * CO + gn] = v;
      }
    }
  }
}

// ---------------------------------------------------------------------------
// Fused gather + first MLP layer of an SA branch (single batch).
// Row m -> (j=m/K, s=m%K), p=ballidx[j*K+s];
// features = concat(pts[p][0:Cpts], xyz[p]-new_xyz[j]). Gather indices are
// hoisted out of the K loop into LDS (rows are fixed per block).
// ---------------------------------------------------------------------------
__global__ __launch_bounds__(128)
void group_linear_wmma_kernel(const float* __restrict__ pts, int Cpts,
                              const float* __restrict__ xyz,
                              const float* __restrict__ new_xyz,
                              const int* __restrict__ gidx, int K, int Npts,
                              const _Float16* __restrict__ Bp,
                              const float* __restrict__ bias,
                              float* __restrict__ Y, int M, int CO, int NT)
{
  __shared__ _Float16 As[64 * AS_STRIDE];
  __shared__ int pj[64];   // gathered point index per tile row (-1 = pad row)
  __shared__ int nj[64];   // centroid index per tile row
  const int CI    = Cpts + 3;
  const int lane  = threadIdx.x & 31;
  const int wave  = threadIdx.x >> 5;
  const int m0    = blockIdx.x * 64;
  const int nt    = blockIdx.y * 4 + wave;
  const int n0    = nt * 16;
  const int half  = lane >> 4;
  const int nn    = lane & 15;
  const bool nok  = (nt < NT);
  const v16h* BpV = (const v16h*)Bp;

  if (threadIdx.x < 64) {
    int gm = m0 + (int)threadIdx.x;
    int p = -1, j = 0;
    if (gm < M) {
      j = gm / K;
      int s = gm - j * K;
      p = gidx[j * K + s];
      p = p < 0 ? 0 : (p >= Npts ? Npts - 1 : p);
    }
    pj[threadIdx.x] = p;
    nj[threadIdx.x] = j;
  }
  __syncthreads();

  v8f acc[4] = {};
  const int KT = (CI + 31) >> 5;
  const bool cvec = ((Cpts & 3) == 0);

  for (int kt = 0; kt < KT; ++kt) {
    const int k0 = kt << 5;
    if (cvec && (k0 + 32 <= Cpts)) {
      // interior of the contiguous channel region: vectorized gather rows
      for (int q = threadIdx.x; q < 64 * 8; q += 128) {
        int r = q >> 3, cq = (q & 7) << 2;
        int p = pj[r];
        v4h h = {};
        if (p >= 0) {
          const float4 f = *(const float4*)(pts + (size_t)p * Cpts + k0 + cq);
          h.x = (_Float16)f.x; h.y = (_Float16)f.y;
          h.z = (_Float16)f.z; h.w = (_Float16)f.w;
        }
        *(v4h*)&As[r * AS_STRIDE + cq] = h;
      }
    } else {
      for (int e = threadIdx.x; e < 64 * 32; e += 128) {
        int r = e >> 5, c = e & 31;
        int gk = k0 + c;
        int p = pj[r];
        float v = 0.0f;
        if (p >= 0 && gk < CI) {
          if (gk < Cpts) v = pts[(size_t)p * Cpts + gk];
          else {
            int c3 = gk - Cpts;
            v = xyz[p * 3 + c3] - new_xyz[nj[r] * 3 + c3];
          }
        }
        As[r * AS_STRIDE + c] = (_Float16)v;
      }
    }
    __syncthreads();

    v16h b = {};
    if (nok) b = BpV[(size_t)(kt * NT + nt) * 32 + lane];

    v16h a0 = a_frag_from_lds(As, 0, nn, half);
    v16h a1 = a_frag_from_lds(As, 1, nn, half);
    v16h a2 = a_frag_from_lds(As, 2, nn, half);
    v16h a3 = a_frag_from_lds(As, 3, nn, half);
    acc[0] = WMMA_F16(a0, b, acc[0]);
    acc[1] = WMMA_F16(a1, b, acc[1]);
    acc[2] = WMMA_F16(a2, b, acc[2]);
    acc[3] = WMMA_F16(a3, b, acc[3]);
    __syncthreads();
  }

  const int gn = n0 + nn;
  const float bv = (gn < CO) ? bias[gn] : 0.0f;
#pragma unroll
  for (int mt = 0; mt < 4; ++mt) {
#pragma unroll
    for (int r = 0; r < 8; ++r) {
      int gm = m0 + mt * 16 + r + (half << 3);
      if (gm < M && gn < CO) {
        float v = fmaxf(acc[mt][r] + bv, 0.0f); // layer-1 always ReLU
        Y[(size_t)gm * CO + gn] = v;
      }
    }
  }
}

// ---------------------------------------------------------------------------
// Farthest-point sampling. One workgroup per batch; distance array in LDS.
// Matches jax.lax.scan semantics: emit old `far`, min-update, argmax -> next.
// ---------------------------------------------------------------------------
__global__ __launch_bounds__(256)
void fps_kernel(const float* __restrict__ xyz, int N, int npoint,
                int* __restrict__ out_idx)
{
  const float* p = xyz + (size_t)blockIdx.x * N * 3;
  int* oidx = out_idx + blockIdx.x * npoint;
  __shared__ float dist[2048];
  __shared__ float rv[256];
  __shared__ int   ri[256];
  __shared__ int   sfar;
  const int t = threadIdx.x;

  for (int i = t; i < N; i += 256) dist[i] = 1e10f;
  if (t == 0) sfar = 0;
  __syncthreads();

  for (int it = 0; it < npoint; ++it) {
    const int far = sfar;
    const float fx = p[far * 3], fy = p[far * 3 + 1], fz = p[far * 3 + 2];
    float bv = -1.0f; int bi = 0;
    for (int i = t; i < N; i += 256) {
      float dx = p[i * 3] - fx, dy = p[i * 3 + 1] - fy, dz = p[i * 3 + 2] - fz;
      float d  = dx * dx + dy * dy + dz * dz;
      float nd = fminf(dist[i], d);
      dist[i] = nd;
      if (nd > bv) { bv = nd; bi = i; }
    }
    rv[t] = bv; ri[t] = bi;
    __syncthreads();
    for (int s = 128; s > 0; s >>= 1) {
      if (t < s) {
        if (rv[t + s] > rv[t] || (rv[t + s] == rv[t] && ri[t + s] < ri[t])) {
          rv[t] = rv[t + s]; ri[t] = ri[t + s];
        }
      }
      __syncthreads();
    }
    if (t == 0) { oidx[it] = far; sfar = ri[0]; }
    __syncthreads();
  }
}

__global__ void gather_xyz_kernel(const float* __restrict__ xyz,
                                  const int* __restrict__ idx,
                                  float* __restrict__ out, int B, int N, int S)
{
  int i = blockIdx.x * blockDim.x + threadIdx.x;
  if (i >= B * S * 3) return;
  int c = i % 3, tj = i / 3, b = tj / S, j = tj - b * S;
  int p = idx[b * S + j];
  p = p < 0 ? 0 : (p >= N ? N - 1 : p);
  out[i] = xyz[((size_t)b * N + p) * 3 + c];
}

// Ball query: first K in-scan-order points with d<=r^2, padded with first hit.
__global__ void ball_query_kernel(const float* __restrict__ xyz,
                                  const float* __restrict__ new_xyz,
                                  int N, int S, float r2, int K,
                                  int* __restrict__ out)
{
  int j = blockIdx.x * blockDim.x + threadIdx.x;
  if (j >= S) return;
  const float* px = xyz + (size_t)blockIdx.y * N * 3;
  const float* pc = new_xyz + (size_t)blockIdx.y * S * 3;
  int* po = out + (size_t)blockIdx.y * S * K;
  float cx = pc[j * 3], cy = pc[j * 3 + 1], cz = pc[j * 3 + 2];
  int cnt = 0, first = N - 1;
  bool found = false;
  for (int p = 0; p < N && cnt < K; ++p) {
    float dx = px[p * 3] - cx, dy = px[p * 3 + 1] - cy, dz = px[p * 3 + 2] - cz;
    float d = dx * dx + dy * dy + dz * dz;
    if (d <= r2) {
      if (!found) { found = true; first = p; }
      po[(size_t)j * K + cnt++] = p;
    }
  }
  for (; cnt < K; ++cnt) po[(size_t)j * K + cnt] = first;
}

__global__ void maxpool_kernel(const float* __restrict__ act,
                               float* __restrict__ out,
                               int npoint, int K, int CO, int out_stride, int co_off)
{
  int i = blockIdx.x * blockDim.x + threadIdx.x;
  if (i >= npoint * CO) return;
  int j = i / CO, c = i - j * CO;
  float m = -1e30f;
  for (int s = 0; s < K; ++s)
    m = fmaxf(m, act[((size_t)j * K + s) * CO + c]);
  out[(size_t)j * out_stride + co_off + c] = m;
}

__global__ void knn3_kernel(const float* __restrict__ xyz1,
                            const float* __restrict__ xyz2,
                            int B, int N1, int S,
                            int* __restrict__ idx, float* __restrict__ w)
{
  int i = blockIdx.x * blockDim.x + threadIdx.x;
  if (i >= B * N1) return;
  int b = i / N1, n = i - b * N1;
  const float* p1 = xyz1 + ((size_t)b * N1 + n) * 3;
  const float* p2 = xyz2 + (size_t)b * S * 3;
  float d0 = 1e30f, d1 = 1e30f, d2 = 1e30f;
  int i0 = 0, i1 = 0, i2 = 0;
  float ax = p1[0], ay = p1[1], az = p1[2];
  for (int s = 0; s < S; ++s) {
    float dx = ax - p2[s * 3], dy = ay - p2[s * 3 + 1], dz = az - p2[s * 3 + 2];
    float d = dx * dx + dy * dy + dz * dz;
    if (d < d0)      { d2 = d1; i2 = i1; d1 = d0; i1 = i0; d0 = d; i0 = s; }
    else if (d < d1) { d2 = d1; i2 = i1; d1 = d; i1 = s; }
    else if (d < d2) { d2 = d; i2 = s; }
  }
  float w0 = 1.0f / (d0 + 1e-8f), w1 = 1.0f / (d1 + 1e-8f), w2 = 1.0f / (d2 + 1e-8f);
  float ws = w0 + w1 + w2;
  idx[i * 3] = i0; idx[i * 3 + 1] = i1; idx[i * 3 + 2] = i2;
  w[i * 3] = w0 / ws; w[i * 3 + 1] = w1 / ws; w[i * 3 + 2] = w2 / ws;
}

__global__ void fp_concat_kernel(const float* __restrict__ points1, int C1,
                                 const float* __restrict__ points2, int C2, int S,
                                 const int* __restrict__ idx,
                                 const float* __restrict__ w,
                                 float* __restrict__ X, int B, int N1)
{
  int CI = C1 + C2;
  int i = blockIdx.x * blockDim.x + threadIdx.x;
  if (i >= B * N1 * CI) return;
  int m = i / CI, c = i - m * CI;
  int b = m / N1;
  float v;
  if (c < C1) v = points1[(size_t)m * C1 + c];
  else {
    int cc = c - C1;
    const float* p2 = points2 + (size_t)b * S * C2;
    const int* id = idx + m * 3;
    const float* ww = w + m * 3;
    v = ww[0] * p2[(size_t)id[0] * C2 + cc] +
        ww[1] * p2[(size_t)id[1] * C2 + cc] +
        ww[2] * p2[(size_t)id[2] * C2 + cc];
  }
  X[i] = v;
}

__global__ void sa3_concat_kernel(const float* __restrict__ l2xyz,
                                  const float* __restrict__ l2pts,
                                  float* __restrict__ X, int rows)
{
  int i = blockIdx.x * blockDim.x + threadIdx.x;
  if (i >= rows * 515) return;
  int m = i / 515, c = i - m * 515;
  X[i] = (c < 3) ? l2xyz[m * 3 + c] : l2pts[(size_t)m * 512 + (c - 3)];
}

__global__ void fp3_concat_kernel(const float* __restrict__ l2pts,
                                  const float* __restrict__ l3,
                                  float* __restrict__ X, int rows)
{
  int i = blockIdx.x * blockDim.x + threadIdx.x;
  if (i >= rows * 1536) return;
  int m = i / 1536, c = i - m * 1536;
  int b = m / 128;
  X[i] = (c < 512) ? l2pts[(size_t)m * 512 + c] : l3[b * 1024 + (c - 512)];
}

__global__ void dup6_kernel(const float* __restrict__ xyz, float* __restrict__ out, int n)
{
  int i = blockIdx.x * blockDim.x + threadIdx.x;
  if (i >= n * 6) return;
  int m = i / 6, c = i - m * 6;
  out[i] = xyz[m * 3 + (c % 3)];
}

// ---------------------------------------------------------------------------
// Host orchestration
// ---------------------------------------------------------------------------
static inline size_t al256(size_t x) { return (x + 255) & ~(size_t)255; }

extern "C" void kernel_launch(void* const* d_in, const int* in_sizes, int n_in,
                              void* d_out, int out_size, void* d_ws, size_t ws_size,
                              hipStream_t stream)
{
  (void)n_in; (void)out_size; (void)ws_size;
  const int B = 8, N0 = 2048;

  // --- input mapping: handle both insertion-order and sorted-key flattening
  int map[51];
  if (in_sizes[0] == B * N0 * 3) {
    for (int i = 0; i < 51; ++i) map[i] = i;           // insertion order
  } else {                                             // sorted pytree order
    map[0] = 50;
    for (int i = 0; i < 18; ++i) map[1 + i]  = 14 + i; // sa1
    for (int i = 0; i < 12; ++i) map[19 + i] = 32 + i; // sa2
    for (int i = 0; i < 6;  ++i) map[31 + i] = 44 + i; // sa3
    for (int i = 0; i < 4;  ++i) map[37 + i] = 10 + i; // fp3
    for (int i = 0; i < 4;  ++i) map[41 + i] = 6 + i;  // fp2
    for (int i = 0; i < 4;  ++i) map[45 + i] = 2 + i;  // fp1
    map[49] = 0; map[50] = 1;                          // conv1
  }
  auto IN = [&](int li) { return (const float*)d_in[map[li]]; };
  const int L_XYZ = 0, L_SA1 = 1, L_SA2 = 19, L_SA3 = 31,
            L_FP3 = 37, L_FP2 = 41, L_FP1 = 45, L_CONV1 = 49;

  const float* xyz = IN(L_XYZ);

  // --- workspace bump allocator --------------------------------------------
  char* wp = (char*)d_ws;
  auto alloc = [&](size_t bytes) { char* r = wp; wp += al256(bytes); return r; };

  float* l1_xyz  = (float*)alloc((size_t)B * 512 * 3 * 4);
  float* l2_xyz  = (float*)alloc((size_t)B * 128 * 3 * 4);
  float* l1_pts  = (float*)alloc((size_t)B * 512 * 320 * 4);
  float* l2_pts  = (float*)alloc((size_t)B * 128 * 512 * 4);
  float* l3_pts  = (float*)alloc((size_t)B * 1024 * 4);
  float* fp3_out = (float*)alloc((size_t)B * 128 * 256 * 4);
  float* fp2_out = (float*)alloc((size_t)B * 512 * 128 * 4);
  float* fp1_out = (float*)alloc((size_t)B * 2048 * 128 * 4);
  float* p1_6    = (float*)alloc((size_t)B * 2048 * 6 * 4);
  float* scrA    = (float*)alloc((size_t)9000000 * 4);   // per-batch ping
  float* scrB    = (float*)alloc((size_t)7000000 * 4);   // per-batch pong
  int*   fidx1   = (int*)alloc((size_t)B * 512 * 4);
  int*   fidx2   = (int*)alloc((size_t)B * 128 * 4);
  int*   ballidx = (int*)alloc((size_t)B * 512 * 128 * 4);
  int*   knn_idx = (int*)alloc((size_t)B * 2048 * 3 * 4);
  float* knn_w   = (float*)alloc((size_t)B * 2048 * 3 * 4);

  auto cdiv = [](int a, int b) { return (a + b - 1) / b; };

  // --- pre-pack all weight matrices into WMMA B-fragment layout (f16) -----
  auto packW = [&](int li, int CO, int CI) -> const _Float16* {
    int NT = cdiv(CO, 16), KT = cdiv(CI, 32);
    size_t total = (size_t)NT * KT * 512;
    _Float16* bp = (_Float16*)alloc(total * 2);
    pack_w_kernel<<<cdiv((int)total, 256), 256, 0, stream>>>(IN(li), bp, CO, CI, NT, KT);
    return bp;
  };

  const int sa1_w[3][3] = {{32, 32, 64}, {64, 64, 128}, {64, 96, 128}};
  const int sa2_w[2][3] = {{128, 128, 256}, {128, 196, 256}};
  const _Float16* Bp_sa1[3][3];
  const _Float16* Bp_sa2[2][3];
  for (int br = 0; br < 3; ++br) {
    int ci = 6;
    for (int l = 0; l < 3; ++l) {
      Bp_sa1[br][l] = packW(L_SA1 + br * 6 + l * 2, sa1_w[br][l], ci);
      ci = sa1_w[br][l];
    }
  }
  for (int br = 0; br < 2; ++br) {
    int ci = 323;
    for (int l = 0; l < 3; ++l) {
      Bp_sa2[br][l] = packW(L_SA2 + br * 6 + l * 2, sa2_w[br][l], ci);
      ci = sa2_w[br][l];
    }
  }
  const _Float16* Bp_sa3[3] = { packW(L_SA3 + 0, 256, 515),
                                packW(L_SA3 + 2, 512, 256),
                                packW(L_SA3 + 4, 1024, 512) };
  const _Float16* Bp_fp3[2] = { packW(L_FP3 + 0, 256, 1536),
                                packW(L_FP3 + 2, 256, 256) };
  const _Float16* Bp_fp2[2] = { packW(L_FP2 + 0, 256, 576),
                                packW(L_FP2 + 2, 128, 256) };
  const _Float16* Bp_fp1[2] = { packW(L_FP1 + 0, 128, 134),
                                packW(L_FP1 + 2, 128, 128) };
  const _Float16* Bp_conv1  =   packW(L_CONV1 + 0, 256, 128);

  auto linear = [&](const float* X, const _Float16* Bp, const float* bias, float* Y,
                    int M, int CI, int CO, int relu) {
    int NT = cdiv(CO, 16);
    linear_wmma_kernel<<<dim3(cdiv(M, 64), cdiv(CO, 64)), 128, 0, stream>>>(
        X, Bp, bias, Y, M, CI, CO, NT, relu);
  };

  // ======================= SA1 (MSG, 2048 -> 512) ==========================
  fps_kernel<<<B, 256, 0, stream>>>(xyz, N0, 512, fidx1);
  gather_xyz_kernel<<<cdiv(B * 512 * 3, 256), 256, 0, stream>>>(xyz, fidx1, l1_xyz, B, N0, 512);

  const int   sa1_K[3]   = {32, 64, 128};
  const float sa1_r[3]   = {0.1f, 0.2f, 0.4f};
  const int   sa1_off[3] = {0, 64, 192};

  for (int br = 0; br < 3; ++br) {
    const int K = sa1_K[br];
    const float r2 = sa1_r[br] * sa1_r[br];
    ball_query_kernel<<<dim3(cdiv(512, 64), B), 64, 0, stream>>>(
        xyz, l1_xyz, N0, 512, r2, K, ballidx);
    const float* b0 = IN(L_SA1 + br * 6 + 1);
    const float* b1 = IN(L_SA1 + br * 6 + 3);
    const float* b2 = IN(L_SA1 + br * 6 + 5);
    const int w0 = sa1_w[br][0], w1 = sa1_w[br][1], w2 = sa1_w[br][2];
    const int M = 512 * K;
    for (int b = 0; b < B; ++b) {
      group_linear_wmma_kernel<<<dim3(cdiv(M, 64), cdiv(w0, 64)), 128, 0, stream>>>(
          xyz + (size_t)b * N0 * 3, 3, xyz + (size_t)b * N0 * 3,
          l1_xyz + (size_t)b * 512 * 3, ballidx + (size_t)b * 512 * K, K, N0,
          Bp_sa1[br][0], b0, scrA, M, w0, cdiv(w0, 16));
      linear(scrA, Bp_sa1[br][1], b1, scrB, M, w0, w1, 1);
      linear(scrB, Bp_sa1[br][2], b2, scrA, M, w1, w2, 1);
      maxpool_kernel<<<cdiv(512 * w2, 256), 256, 0, stream>>>(
          scrA, l1_pts + (size_t)b * 512 * 320, 512, K, w2, 320, sa1_off[br]);
    }
  }

  // ======================= SA2 (MSG, 512 -> 128) ===========================
  fps_kernel<<<B, 256, 0, stream>>>(l1_xyz, 512, 128, fidx2);
  gather_xyz_kernel<<<cdiv(B * 128 * 3, 256), 256, 0, stream>>>(l1_xyz, fidx2, l2_xyz, B, 512, 128);

  const int   sa2_K[2]   = {64, 128};
  const float sa2_r[2]   = {0.4f, 0.8f};
  const int   sa2_off[2] = {0, 256};

  for (int br = 0; br < 2; ++br) {
    const int K = sa2_K[br];
    const float r2 = sa2_r[br] * sa2_r[br];
    ball_query_kernel<<<dim3(cdiv(128, 64), B), 64, 0, stream>>>(
        l1_xyz, l2_xyz, 512, 128, r2, K, ballidx);
    const float* b0 = IN(L_SA2 + br * 6 + 1);
    const float* b1 = IN(L_SA2 + br * 6 + 3);
    const float* b2 = IN(L_SA2 + br * 6 + 5);
    const int w0 = sa2_w[br][0], w1 = sa2_w[br][1], w2 = sa2_w[br][2];
    const int M = 128 * K;
    for (int b = 0; b < B; ++b) {
      group_linear_wmma_kernel<<<dim3(cdiv(M, 64), cdiv(w0, 64)), 128, 0, stream>>>(
          l1_pts + (size_t)b * 512 * 320, 320, l1_xyz + (size_t)b * 512 * 3,
          l2_xyz + (size_t)b * 128 * 3, ballidx + (size_t)b * 128 * K, K, 512,
          Bp_sa2[br][0], b0, scrA, M, w0, cdiv(w0, 16));
      linear(scrA, Bp_sa2[br][1], b1, scrB, M, w0, w1, 1);
      linear(scrB, Bp_sa2[br][2], b2, scrA, M, w1, w2, 1);
      maxpool_kernel<<<cdiv(128 * w2, 256), 256, 0, stream>>>(
          scrA, l2_pts + (size_t)b * 128 * 512, 128, K, w2, 512, sa2_off[br]);
    }
  }

  // ======================= SA3 (group-all) =================================
  {
    const int rows = B * 128;
    sa3_concat_kernel<<<cdiv(rows * 515, 256), 256, 0, stream>>>(l2_xyz, l2_pts, scrA, rows);
    linear(scrA, Bp_sa3[0], IN(L_SA3 + 1), scrB, rows, 515, 256, 1);
    linear(scrB, Bp_sa3[1], IN(L_SA3 + 3), scrA, rows, 256, 512, 1);
    linear(scrA, Bp_sa3[2], IN(L_SA3 + 5), scrB, rows, 512, 1024, 1);
    maxpool_kernel<<<cdiv(B * 1024, 256), 256, 0, stream>>>(scrB, l3_pts, B, 128, 1024, 1024, 0);
  }

  // ======================= FP3 (l3 -> l2, broadcast) =======================
  {
    const int rows = B * 128;
    fp3_concat_kernel<<<cdiv(rows * 1536, 256), 256, 0, stream>>>(l2_pts, l3_pts, scrA, rows);
    linear(scrA, Bp_fp3[0], IN(L_FP3 + 1), scrB, rows, 1536, 256, 1);
    linear(scrB, Bp_fp3[1], IN(L_FP3 + 3), fp3_out, rows, 256, 256, 1);
  }

  // ======================= FP2 (l2 -> l1, 3-NN) ============================
  {
    const int rows = B * 512;
    knn3_kernel<<<cdiv(rows, 128), 128, 0, stream>>>(l1_xyz, l2_xyz, B, 512, 128, knn_idx, knn_w);
    fp_concat_kernel<<<cdiv(rows * 576, 256), 256, 0, stream>>>(
        l1_pts, 320, fp3_out, 256, 128, knn_idx, knn_w, scrA, B, 512);
    linear(scrA, Bp_fp2[0], IN(L_FP2 + 1), scrB, rows, 576, 256, 1);
    linear(scrB, Bp_fp2[1], IN(L_FP2 + 3), fp2_out, rows, 256, 128, 1);
  }

  // ======================= FP1 (l1 -> l0, 3-NN) ============================
  {
    const int rows = B * 2048;
    dup6_kernel<<<cdiv(rows * 6, 256), 256, 0, stream>>>(xyz, p1_6, rows);
    knn3_kernel<<<cdiv(rows, 128), 128, 0, stream>>>(xyz, l1_xyz, B, 2048, 512, knn_idx, knn_w);
    fp_concat_kernel<<<cdiv(rows * 134, 256), 256, 0, stream>>>(
        p1_6, 6, fp2_out, 128, 512, knn_idx, knn_w, scrA, B, 2048);
    linear(scrA, Bp_fp1[0], IN(L_FP1 + 1), scrB, rows, 134, 128, 1);
    linear(scrB, Bp_fp1[1], IN(L_FP1 + 3), fp1_out, rows, 128, 128, 1);
    // final conv1: 128 -> 256, bias, NO relu, straight into d_out
    linear(fp1_out, Bp_conv1, IN(L_CONV1 + 1), (float*)d_out, rows, 128, 256, 0);
  }
}
